// ChunkedSparseAttention_3899830305036
// MI455X (gfx1250) — compile-verified, matrix operand order checked by reading, not run on hardware
//
#include <hip/hip_runtime.h>
#include <hip/hip_bf16.h>

// CDNA5 / gfx1250 chunked sparse attention (local + prefix flash attention).
// bf16 WMMA (v_wmma_f32_16x16x32_bf16) + async global->LDS K staging +
// pre-packed (bf16, transposed) V streamed from L2 via global_load_b128.

typedef __attribute__((ext_vector_type(16))) __bf16 v16bf;
typedef __attribute__((ext_vector_type(8)))  float  v8f;
typedef __attribute__((ext_vector_type(4)))  int    v4i;

// address-space-qualified int4 pointers for the async global->LDS builtin
typedef __attribute__((address_space(1))) v4i* gv4i_p;
typedef __attribute__((address_space(3))) v4i* lv4i_p;

#define S_LEN    4096
#define D_DIM    1024
#define CHUNK_SZ 64
#define NCH      (S_LEN / CHUNK_SZ)
#define ALPHA    0.1f
#define NTHREADS 512          // 16 wave32 waves
#define SLICE_D  512          // K chunk staged as two d-slices (double buffer)

// padded LDS row strides (in bf16 elems) to spread lanes across banks
#define LQ_LD 1032
#define LK_LD 520
#define LP_LD 72

// ---- dynamic LDS layout (bytes) ----
#define OFF_Q     0
#define OFF_K0    (CHUNK_SZ * LQ_LD * 2)            // 132096
#define OFF_K1    (OFF_K0 + CHUNK_SZ * LK_LD * 2)   // 198656
#define OFF_S     (OFF_K1 + CHUNK_SZ * LK_LD * 2)   // 265216
#define OFF_P     (OFF_S + CHUNK_SZ * CHUNK_SZ * 4) // 281600
#define OFF_M     (OFF_P + CHUNK_SZ * LP_LD * 2)    // 290816
#define OFF_L     (OFF_M + 256)
#define OFF_SC    (OFF_L + 256)
#define LDS_BYTES (OFF_SC + 256)                    // 291584 < 320KB

#if __has_builtin(__builtin_amdgcn_global_load_async_to_lds_b128)
#define HAVE_ASYNC 1
#else
#define HAVE_ASYNC 0
#endif

template <int N>
__device__ __forceinline__ void wait_async() {
#if HAVE_ASYNC
#if __has_builtin(__builtin_amdgcn_s_wait_asynccnt)
  __builtin_amdgcn_s_wait_asynccnt(N);
#else
  asm volatile("s_wait_asynccnt %0" ::"i"(N) : "memory");
#endif
#endif
}

// ---------------------------------------------------------------------------
// bf16 fragment load, CDNA5 ISA 7.12.2 layout (wave32): lane%16 = row;
// lanes 0-15 hold K 0..7 (v0-3) + 16..23 (v4-7); lanes 16-31 hold K 8..15 +
// 24..31. Two contiguous 16B segments per lane -> ds/global_load_b128 pairs.
// Works for LDS or global source pointers.
// ---------------------------------------------------------------------------
__device__ __forceinline__ v16bf load_frag_rowmajor(const __bf16* src, int ld,
                                                    int rb, int kb) {
  const int lane = threadIdx.x & 31;
  const int g    = lane >> 4;
  const __bf16* p = src + (size_t)(rb + (lane & 15)) * ld + kb + (g << 3);
  v16bf f;
#pragma unroll
  for (int v = 0; v < 8; ++v) {
    const int k0 = ((v & 3) << 1) + ((v >> 2) << 4);
    f[2 * v]     = p[k0];
    f[2 * v + 1] = p[k0 + 1];
  }
  return f;
}

// fallback: V fragment straight from f32 k-major global memory (+cvt)
__device__ __forceinline__ v16bf load_frag_v_f32(const float* v, int kb,
                                                 int nb) {
  const int lane = threadIdx.x & 31;
  const int g    = lane >> 4;
  const float* p = v + (size_t)(kb + (g << 3)) * D_DIM + nb + (lane & 15);
  v16bf f;
#pragma unroll
  for (int q = 0; q < 8; ++q) {
    const int k0 = ((q & 3) << 1) + ((q >> 2) << 4);
    f[2 * q]     = (__bf16)p[(size_t)k0 * D_DIM];
    f[2 * q + 1] = (__bf16)p[(size_t)(k0 + 1) * D_DIM];
  }
  return f;
}

__device__ __forceinline__ v8f wmma_bf16(v16bf a, v16bf b, v8f c) {
  return __builtin_amdgcn_wmma_f32_16x16x32_bf16(
      false, a, false, b, (short)0, c, false, false);
}

// O[16 x 256 per wave] += P x V, 2-stage software pipeline: fetch tile t+1's
// B fragments between tile t's WMMAs so loads overlap matrix math.
template <bool PACKED>
__device__ __forceinline__ void pv_accum(v16bf a0, v16bf a1,
                                         const __bf16* vtc, const float* vc,
                                         int cb, v8f* acc) {
  v16bf b0, b1;
  if (PACKED) {
    b0 = load_frag_rowmajor(vtc, CHUNK_SZ, cb, 0);
    b1 = load_frag_rowmajor(vtc, CHUNK_SZ, cb, 32);
  } else {
    b0 = load_frag_v_f32(vc, 0, cb);
    b1 = load_frag_v_f32(vc, 32, cb);
  }
#pragma unroll
  for (int t = 0; t < 16; ++t) {
    v16bf n0 = b0, n1 = b1;
    acc[t] = wmma_bf16(a0, b0, acc[t]);
    if (t < 15) {
      const int nb = cb + ((t + 1) << 4);
      if (PACKED) {
        n0 = load_frag_rowmajor(vtc, CHUNK_SZ, nb, 0);
        n1 = load_frag_rowmajor(vtc, CHUNK_SZ, nb, 32);
      } else {
        n0 = load_frag_v_f32(vc, 0, nb);
        n1 = load_frag_v_f32(vc, 32, nb);
      }
    }
    acc[t] = wmma_bf16(a1, b1, acc[t]);
    b0 = n0;
    b1 = n1;
  }
}

// Stage one 64 x 512 K d-slice into (padded) LDS.
// PACKED: raw bf16 copy -> async global->LDS when available.
// !PACKED: f32 load + cvt + ds store.
template <bool PACKED>
__device__ __forceinline__ void issue_k_slice(__bf16* dst, const __bf16* kp,
                                              const float* kf, size_t row0,
                                              int dbase) {
  if (PACKED) {
#if HAVE_ASYNC
    // 4096 x 16B; 8 async instructions per wave (ASYNCcnt += 8)
    for (int i = threadIdx.x; i < (CHUNK_SZ * SLICE_D) / 8; i += NTHREADS) {
      const int row = i >> 6;
      const int u   = (i & 63) << 3;
      const __bf16* gsrc = kp + (row0 + row) * D_DIM + dbase + u;
      __builtin_amdgcn_global_load_async_to_lds_b128(
          (gv4i_p)gsrc, (lv4i_p)(dst + row * LK_LD + u), 0, 0);
    }
#else
    for (int i = threadIdx.x; i < (CHUNK_SZ * SLICE_D) / 8; i += NTHREADS) {
      const int row = i >> 6;
      const int u   = (i & 63) << 3;
      *(uint4*)(dst + row * LK_LD + u) =
          *(const uint4*)(kp + (row0 + row) * D_DIM + dbase + u);
    }
#endif
  } else {
    for (int i = threadIdx.x; i < (CHUNK_SZ * SLICE_D) / 4; i += NTHREADS) {
      const int row = i >> 7;
      const int c   = (i & 127) << 2;
      const float4 x = *(const float4*)(kf + (row0 + row) * D_DIM + dbase + c);
      union { __bf16 h[4]; uint2 u2; } t;
      t.h[0] = (__bf16)x.x; t.h[1] = (__bf16)x.y;
      t.h[2] = (__bf16)x.z; t.h[3] = (__bf16)x.w;
      *(uint2*)(dst + row * LK_LD + c) = t.u2;
    }
  }
}

// online softmax, 8 lanes per row (512 threads = 64 rows)
__device__ __forceinline__ void softmax_online_par(float* ls, __bf16* lp,
                                                   float* lm, float* ll,
                                                   float* lsc) {
  const int r  = threadIdx.x >> 3;
  const int c0 = (threadIdx.x & 7) << 3;
  float s[8];
#pragma unroll
  for (int k = 0; k < 8; ++k) s[k] = ls[r * CHUNK_SZ + c0 + k];
  float mx = s[0];
#pragma unroll
  for (int k = 1; k < 8; ++k) mx = fmaxf(mx, s[k]);
#pragma unroll
  for (int off = 1; off < 8; off <<= 1) mx = fmaxf(mx, __shfl_xor(mx, off, 8));
  const float mold = lm[r];
  mx = fmaxf(mx, mold);
  float sum = 0.f;
#pragma unroll
  for (int k = 0; k < 8; ++k) {
    const float e = __expf(s[k] - mx);
    lp[r * LP_LD + c0 + k] = (__bf16)e;
    sum += e;
  }
#pragma unroll
  for (int off = 1; off < 8; off <<= 1) sum += __shfl_xor(sum, off, 8);
  if ((threadIdx.x & 7) == 0) {
    const float c = __expf(mold - mx);
    lm[r]  = mx;
    ll[r]  = ll[r] * c + sum;
    lsc[r] = c;
  }
}

// exact softmax for the local chunk, scaled by beta
__device__ __forceinline__ void softmax_local_par(float* ls, __bf16* lp,
                                                  float beta) {
  const int r  = threadIdx.x >> 3;
  const int c0 = (threadIdx.x & 7) << 3;
  float s[8];
#pragma unroll
  for (int k = 0; k < 8; ++k) s[k] = ls[r * CHUNK_SZ + c0 + k];
  float mx = s[0];
#pragma unroll
  for (int k = 1; k < 8; ++k) mx = fmaxf(mx, s[k]);
#pragma unroll
  for (int off = 1; off < 8; off <<= 1) mx = fmaxf(mx, __shfl_xor(mx, off, 8));
  float e[8], sum = 0.f;
#pragma unroll
  for (int k = 0; k < 8; ++k) { e[k] = __expf(s[k] - mx); sum += e[k]; }
#pragma unroll
  for (int off = 1; off < 8; off <<= 1) sum += __shfl_xor(sum, off, 8);
  const float inv = beta / sum;
#pragma unroll
  for (int k = 0; k < 8; ++k) lp[r * LP_LD + c0 + k] = (__bf16)(e[k] * inv);
}

template <bool PACKED>
__global__ void __launch_bounds__(NTHREADS, 1)
attn_main(const float* __restrict__ Qf, const float* __restrict__ Kf,
          const float* __restrict__ Vf, const __bf16* __restrict__ Kp,
          const __bf16* __restrict__ Vt, float* __restrict__ O) {
  extern __shared__ char smem[];
  __bf16* lq  = (__bf16*)(smem + OFF_Q);
  __bf16* lk0 = (__bf16*)(smem + OFF_K0);
  __bf16* lk1 = (__bf16*)(smem + OFF_K1);
  float*  ls  = (float*)(smem + OFF_S);
  __bf16* lp  = (__bf16*)(smem + OFF_P);
  float*  lm  = (float*)(smem + OFF_M);
  float*  ll  = (float*)(smem + OFF_L);
  float*  lsc = (float*)(smem + OFF_SC);

  // heaviest chunks (longest prefix) start first to shrink the tail
  const int ci = NCH - 1 - blockIdx.x;
  const int bb = blockIdx.y;
  const float scale = 0.03125f;               // 1/sqrt(1024), folded into Q
  const size_t rowQ = (size_t)bb * S_LEN + (size_t)ci * CHUNK_SZ;
  const size_t kv_row0 = (size_t)bb * S_LEN;

  // Q -> LDS bf16, pre-scaled, padded stride
  for (int i = threadIdx.x; i < (CHUNK_SZ * D_DIM) / 4; i += NTHREADS) {
    const int row = i >> 8, c = (i & 255) << 2;
    const float4 x = *(const float4*)(Qf + (rowQ + row) * D_DIM + c);
    union { __bf16 h[4]; uint2 u2; } t;
    t.h[0] = (__bf16)(x.x * scale); t.h[1] = (__bf16)(x.y * scale);
    t.h[2] = (__bf16)(x.z * scale); t.h[3] = (__bf16)(x.w * scale);
    *(uint2*)(lq + row * LQ_LD + c) = t.u2;
  }
  if (threadIdx.x < CHUNK_SZ) {
    lm[threadIdx.x] = -3.0e38f;
    ll[threadIdx.x] = 0.f;
  }
  if (ci > 0) issue_k_slice<PACKED>(lk0, Kp, Kf, kv_row0, 0);  // prologue

  const int lane = threadIdx.x & 31, w = threadIdx.x >> 5;
  const int tr  = (w >> 2) << 4;              // row-tile base (all phases)
  const int stc = (w & 3) << 4;               // score col-tile base
  const int cb  = (w & 3) << 8;               // 256 output cols per wave
  const int g   = lane >> 4;

  v8f acc[16];
#pragma unroll
  for (int t = 0; t < 16; ++t)
    acc[t] = (v8f){0.f, 0.f, 0.f, 0.f, 0.f, 0.f, 0.f, 0.f};

  __syncthreads();

  // ---------------- cross (growing-prefix) flash attention ----------------
  for (int j = 0; j < ci; ++j) {
    const size_t row0 = kv_row0 + (size_t)j * CHUNK_SZ;
    v8f sacc = (v8f){0.f, 0.f, 0.f, 0.f, 0.f, 0.f, 0.f, 0.f};

    // slice 0 compute, slice 1 in flight
    issue_k_slice<PACKED>(lk1, Kp, Kf, row0, SLICE_D);
    if (PACKED)
      __builtin_prefetch(Vt + (size_t)(bb * NCH + j) * D_DIM * CHUNK_SZ +
                             (size_t)threadIdx.x * 128, 0, 1);
    else
      __builtin_prefetch(Vf + row0 * D_DIM + (size_t)threadIdx.x * 64, 0, 1);
    wait_async<8>();
    __syncthreads();
#pragma unroll 4
    for (int k = 0; k < SLICE_D; k += 32) {
      v16bf a = load_frag_rowmajor(lq, LQ_LD, tr, k);
      v16bf b = load_frag_rowmajor(lk0, LK_LD, stc, k);  // K^T: row = key idx
      sacc = wmma_bf16(a, b, sacc);
    }
    __syncthreads();

    // slice 1 compute, next chunk's slice 0 in flight
    if (j + 1 < ci) {
      issue_k_slice<PACKED>(lk0, Kp, Kf, row0 + CHUNK_SZ, 0);
      wait_async<8>();
    } else {
      wait_async<0>();
    }
    __syncthreads();
#pragma unroll 4
    for (int k = 0; k < SLICE_D; k += 32) {
      v16bf a = load_frag_rowmajor(lq, LQ_LD, tr, SLICE_D + k);
      v16bf b = load_frag_rowmajor(lk1, LK_LD, stc, k);
      sacc = wmma_bf16(a, b, sacc);
    }
    {
      const int n = lane & 15;
#pragma unroll
      for (int v = 0; v < 8; ++v)
        ls[(tr + v + (g << 3)) * CHUNK_SZ + stc + n] = sacc[v];
    }
    __syncthreads();
    softmax_online_par(ls, lp, lm, ll, lsc);
    __syncthreads();

    // rescale running accumulator, then O += P x V (V streamed from global)
    float f8[8];
#pragma unroll
    for (int v = 0; v < 8; ++v) f8[v] = lsc[tr + v + (g << 3)];
#pragma unroll
    for (int t = 0; t < 16; ++t)
#pragma unroll
      for (int v = 0; v < 8; ++v) acc[t][v] *= f8[v];

    v16bf a0 = load_frag_rowmajor(lp, LP_LD, tr, 0);
    v16bf a1 = load_frag_rowmajor(lp, LP_LD, tr, 32);
    const __bf16* vtc = Vt + (size_t)(bb * NCH + j) * D_DIM * CHUNK_SZ;
    const float*  vc  = Vf + row0 * D_DIM;
    pv_accum<PACKED>(a0, a1, vtc, vc, cb, acc);
  }

  // ---------------- normalize cross part: acc *= ALPHA / l ----------------
  __syncthreads();
  if (ci > 0) {
    if (threadIdx.x < CHUNK_SZ) lsc[threadIdx.x] = ALPHA / ll[threadIdx.x];
    __syncthreads();
    float f8[8];
#pragma unroll
    for (int v = 0; v < 8; ++v) f8[v] = lsc[tr + v + (g << 3)];
#pragma unroll
    for (int t = 0; t < 16; ++t)
#pragma unroll
      for (int v = 0; v < 8; ++v) acc[t][v] *= f8[v];
  }

  // ---------------- local (intra-chunk) attention, exact softmax ----------
  issue_k_slice<PACKED>(lk0, Kp, Kf, rowQ, 0);
  issue_k_slice<PACKED>(lk1, Kp, Kf, rowQ, SLICE_D);
  wait_async<0>();
  __syncthreads();
  {
    v8f sacc = (v8f){0.f, 0.f, 0.f, 0.f, 0.f, 0.f, 0.f, 0.f};
#pragma unroll 4
    for (int k = 0; k < SLICE_D; k += 32) {
      v16bf a = load_frag_rowmajor(lq, LQ_LD, tr, k);
      v16bf b = load_frag_rowmajor(lk0, LK_LD, stc, k);
      sacc = wmma_bf16(a, b, sacc);
    }
#pragma unroll 4
    for (int k = 0; k < SLICE_D; k += 32) {
      v16bf a = load_frag_rowmajor(lq, LQ_LD, tr, SLICE_D + k);
      v16bf b = load_frag_rowmajor(lk1, LK_LD, stc, k);
      sacc = wmma_bf16(a, b, sacc);
    }
    const int n = lane & 15;
#pragma unroll
    for (int v = 0; v < 8; ++v)
      ls[(tr + v + (g << 3)) * CHUNK_SZ + stc + n] = sacc[v];
  }
  __syncthreads();
  softmax_local_par(ls, lp, (ci > 0) ? (1.f - ALPHA) : 1.f);
  __syncthreads();
  {
    v16bf a0 = load_frag_rowmajor(lp, LP_LD, tr, 0);
    v16bf a1 = load_frag_rowmajor(lp, LP_LD, tr, 32);
    const __bf16* vtc = Vt + (size_t)(bb * NCH + ci) * D_DIM * CHUNK_SZ;
    const float*  vc  = Vf + rowQ * D_DIM;
    pv_accum<PACKED>(a0, a1, vtc, vc, cb, acc);
  }

  // ---------------- store f32 output ----------------
  float* outp = O + rowQ * D_DIM;
#pragma unroll
  for (int t = 0; t < 16; ++t) {
    const int nb = cb + (t << 4) + (lane & 15);
#pragma unroll
    for (int v = 0; v < 8; ++v)
      outp[(size_t)(tr + v + (g << 3)) * D_DIM + nb] = acc[t][v];
  }
}

// ---------------- pre-pass: K f32 -> bf16 flat ----------------
__global__ void __launch_bounds__(256)
prepack_k(const float* __restrict__ src, __bf16* __restrict__ dst, int n4) {
  const int i = blockIdx.x * 256 + threadIdx.x;
  if (i < n4) {
    const float4 x = ((const float4*)src)[i];
    union { __bf16 h[4]; uint2 u2; } t;
    t.h[0] = (__bf16)x.x; t.h[1] = (__bf16)x.y;
    t.h[2] = (__bf16)x.z; t.h[3] = (__bf16)x.w;
    ((uint2*)dst)[i] = t.u2;
  }
}

// ---------------- pre-pass: V f32 [b,s,d] -> bf16 [b][chunk][d][64] --------
__global__ void __launch_bounds__(256)
prepack_vt(const float* __restrict__ V, __bf16* __restrict__ Vt) {
  __shared__ float tile[64 * 65];
  const int bc = blockIdx.x;                  // b*NCH + chunk
  const int d0 = blockIdx.y << 6;
  const size_t inbase = (size_t)bc * CHUNK_SZ * D_DIM;
  for (int i = threadIdx.x; i < 64 * 64; i += 256) {
    const int k = i >> 6, dd = i & 63;
    tile[dd * 65 + k] = V[inbase + (size_t)k * D_DIM + d0 + dd];
  }
  __syncthreads();
  const size_t outbase = (size_t)bc * D_DIM * CHUNK_SZ + (size_t)d0 * CHUNK_SZ;
  for (int i = threadIdx.x; i < 64 * 64; i += 256) {
    const int dr = i >> 6, kk = i & 63;
    Vt[outbase + (size_t)dr * CHUNK_SZ + kk] = (__bf16)tile[dr * 65 + kk];
  }
}

extern "C" void kernel_launch(void* const* d_in, const int* in_sizes, int n_in,
                              void* d_out, int out_size, void* d_ws,
                              size_t ws_size, hipStream_t stream) {
  const float* Q = (const float*)d_in[0];
  const float* K = (const float*)d_in[1];
  const float* V = (const float*)d_in[2];
  float*       O = (float*)d_out;

  const int    B      = in_sizes[0] / (S_LEN * D_DIM);   // = 2
  const size_t nelem  = (size_t)B * S_LEN * D_DIM;
  const size_t packed = 2 * nelem * sizeof(__bf16);      // K + Vt in scratch

  dim3 grid(NCH, B);
  if (ws_size >= packed) {
    __bf16* Kp = (__bf16*)d_ws;
    __bf16* Vt = Kp + nelem;
    prepack_k<<<(int)((nelem / 4 + 255) / 256), 256, 0, stream>>>(
        K, Kp, (int)(nelem / 4));
    prepack_vt<<<dim3(B * NCH, D_DIM / 64), 256, 0, stream>>>(V, Vt);
    (void)hipFuncSetAttribute((const void*)attn_main<true>,
                              hipFuncAttributeMaxDynamicSharedMemorySize,
                              LDS_BYTES);
    attn_main<true><<<grid, NTHREADS, LDS_BYTES, stream>>>(Q, K, V, Kp, Vt, O);
  } else {
    (void)hipFuncSetAttribute((const void*)attn_main<false>,
                              hipFuncAttributeMaxDynamicSharedMemorySize,
                              LDS_BYTES);
    attn_main<false><<<grid, NTHREADS, LDS_BYTES, stream>>>(Q, K, V, nullptr,
                                                            nullptr, O);
  }
}
